// GraphBotDetector_42210938585248
// MI455X (gfx1250) — matrix operand order repaired and checked by reading, not compiled, not run on hardware
//
#include <hip/hip_runtime.h>
#include <hip/hip_bf16.h>
#include <stdint.h>

// ---------------------------------------------------------------------------
// GAT layer (PyG GATConv, 3 heads x 128) + global mean pool + linear head
// for MI455X / gfx1250.
//   - h = x@W via v_wmma_f32_16x16x32_bf16, attention logits fused into the
//     GEMM epilogue (shfl reductions over the WMMA C layout).
//   - dst-CSR built once (histogram + LDS scan + scatter), then segment
//     softmax + weighted aggregation as a single atomic-free gather kernel.
// ---------------------------------------------------------------------------

typedef __attribute__((ext_vector_type(16))) __bf16 v16bf;
typedef __attribute__((ext_vector_type(8)))  float  v8f;

#define F_IN        256
#define HEADS       3
#define OUTC        128
#define HO          384      // HEADS*OUTC
#define NUM_GRAPHS  64
#define NUM_CLASSES 2
#define SLOPE       0.2f

// ------------------------------ utility ------------------------------------

__global__ void k_fill_u32(uint32_t* __restrict__ p, uint32_t v, int n) {
  int i = blockIdx.x * blockDim.x + threadIdx.x;
  if (i < n) p[i] = v;
}

__device__ __forceinline__ uint16_t f2bf_bits(float f) {
  __bf16 b = (__bf16)f;
  return __builtin_bit_cast(uint16_t, b);
}

__device__ __forceinline__ float lrelu(float v) { return v > 0.f ? v : SLOPE * v; }

// edge e in [0, E): real edge; e in [E, E+N): self loop (src=dst=e-E)
__device__ __forceinline__ void edge_sd(const int* __restrict__ ei, int e, int E,
                                        int& s, int& d) {
  if (e < E) { s = ei[e]; d = ei[E + e]; }
  else       { s = e - E; d = s; }
}

// -------------------- W pre-swizzle into WMMA B-fragment order --------------
// Wsw[((kt*24 + ct)*32 + lane)*16 + j] =
//     bf16( W[kt*32 + (lane>=16?16:0)+j][ct*16 + (lane&15)] )
__global__ void k_swizzle_W(const float* __restrict__ W, uint16_t* __restrict__ Wsw) {
  int tid = blockIdx.x * blockDim.x + threadIdx.x;   // 8*24*32*16 = 98304 total
  if (tid >= 8 * 24 * 32 * 16) return;
  int j    = tid & 15;
  int rest = tid >> 4;
  int lane = rest & 31; rest >>= 5;
  int ct   = rest % 24;
  int kt   = rest / 24;
  int k    = kt * 32 + ((lane >= 16) ? 16 : 0) + j;
  int col  = ct * 16 + (lane & 15);
  Wsw[tid] = f2bf_bits(W[k * HO + col]);
}

// -------------------- GEMM + fused attention logits -------------------------
// 192 threads = 6 waves: wave = (rowHalf in {0,1}) * 3 + head.
// Each wave computes a 16x128 strip of h (8 WMMA tiles, one full head), then
// a_src[m,hd], a_dst[m,hd] via per-lane dot + 16-lane shfl_xor reduction.
__global__ void __launch_bounds__(192)
k_gemm_wmma(const float* __restrict__ x, const uint16_t* __restrict__ Wsw,
            const float* __restrict__ att_src, const float* __restrict__ att_dst,
            float* __restrict__ h, float* __restrict__ a_src,
            float* __restrict__ a_dst, int n_nodes) {
  const int lane    = threadIdx.x & 31;
  const int wave    = threadIdx.x >> 5;            // 0..5
  const int rowHalf = wave / 3;                    // 0..1
  const int hd      = wave - rowHalf * 3;          // head 0..2
  const int rowbase = (blockIdx.x * 2 + rowHalf) * 16;

  int mA = rowbase + (lane & 15);
  int mclamp = (mA < n_nodes) ? mA : (n_nodes - 1);
  const int kb = (lane >> 4) ? 16 : 0;             // lanes 16-31 -> K base +16
  const float* xrow = x + (size_t)mclamp * F_IN + kb;

  v8f acc[8] = {};
  const uint16_t* wb0 = Wsw + ((size_t)(hd * 8) * 32 + lane) * 16;

  for (int kt = 0; kt < 8; ++kt) {
    // A fragment: K offsets {0..15} relative to kb
    const float4 f0 = *(const float4*)(xrow + kt * 32 + 0);
    const float4 f1 = *(const float4*)(xrow + kt * 32 + 4);
    const float4 f2 = *(const float4*)(xrow + kt * 32 + 8);
    const float4 f3 = *(const float4*)(xrow + kt * 32 + 12);
    v16bf A;
    A[0]  = (__bf16)f0.x; A[1]  = (__bf16)f0.y; A[2]  = (__bf16)f0.z; A[3]  = (__bf16)f0.w;
    A[4]  = (__bf16)f1.x; A[5]  = (__bf16)f1.y; A[6]  = (__bf16)f1.z; A[7]  = (__bf16)f1.w;
    A[8]  = (__bf16)f2.x; A[9]  = (__bf16)f2.y; A[10] = (__bf16)f2.z; A[11] = (__bf16)f2.w;
    A[12] = (__bf16)f3.x; A[13] = (__bf16)f3.y; A[14] = (__bf16)f3.z; A[15] = (__bf16)f3.w;

    const uint16_t* wb = wb0 + (size_t)kt * 24 * 32 * 16;
#pragma unroll
    for (int t = 0; t < 8; ++t) {
      v16bf B = *reinterpret_cast<const v16bf*>(wb + (size_t)t * 32 * 16);
      acc[t] = __builtin_amdgcn_wmma_f32_16x16x32_bf16(
          false, A, false, B, (short)0, acc[t], false, false);
    }
  }

  // C/D layout: VGPR r -> M = r + (lane>=16 ? 8 : 0), N = lane&15
  const int mstorebase = rowbase + ((lane >> 4) ? 8 : 0);
  const int c0 = lane & 15;                        // channel offset within tile

  // store h strip
#pragma unroll
  for (int t = 0; t < 8; ++t) {
#pragma unroll
    for (int r = 0; r < 8; ++r) {
      int m = mstorebase + r;
      if (m < n_nodes) h[(size_t)m * HO + hd * OUTC + t * 16 + c0] = acc[t][r];
    }
  }

  // fused logits: a_src[m,hd] = sum_c h[m][hd*128+c] * att_src[hd][c]
  float ws[8], wd[8];
#pragma unroll
  for (int t = 0; t < 8; ++t) {
    ws[t] = att_src[hd * OUTC + t * 16 + c0];
    wd[t] = att_dst[hd * OUTC + t * 16 + c0];
  }
#pragma unroll
  for (int r = 0; r < 8; ++r) {
    float ps = 0.f, pd = 0.f;
#pragma unroll
    for (int t = 0; t < 8; ++t) { ps += acc[t][r] * ws[t]; pd += acc[t][r] * wd[t]; }
    // reduce across the 16 lanes of this half-wave (masks < 16 stay in half)
#pragma unroll
    for (int mask = 1; mask < 16; mask <<= 1) {
      ps += __shfl_xor(ps, mask);
      pd += __shfl_xor(pd, mask);
    }
    int m = mstorebase + r;
    if (((lane & 15) == 0) && m < n_nodes) {
      a_src[m * HEADS + hd] = ps;
      a_dst[m * HEADS + hd] = pd;
    }
  }
}

// ------------------------- CSR build (dst-grouped) --------------------------
__global__ void k_deg(const int* __restrict__ ei, int* __restrict__ deg,
                      int E, int total) {
  int e = blockIdx.x * blockDim.x + threadIdx.x;
  if (e >= total) return;
  int s, d; edge_sd(ei, e, E, s, d);
  atomicAdd(&deg[d], 1);
}

// exclusive prefix sum of deg -> rowstart[0..n] and cursor copy (1024 threads)
__global__ void __launch_bounds__(1024)
k_scan(const int* __restrict__ deg, int* __restrict__ rowstart,
       int* __restrict__ cursor, int n) {
  __shared__ int sh[1024];
  const int t = threadIdx.x;
  const int chunk = (n + 1023) / 1024;
  const int lo = t * chunk;
  const int hi = (lo + chunk < n) ? (lo + chunk) : n;
  int sum = 0;
  for (int i = lo; i < hi; ++i) sum += deg[i];
  sh[t] = sum;
  __syncthreads();
  for (int off = 1; off < 1024; off <<= 1) {
    int v = (t >= off) ? sh[t - off] : 0;
    __syncthreads();
    sh[t] += v;
    __syncthreads();
  }
  int run = (t == 0) ? 0 : sh[t - 1];              // exclusive prefix of chunk
  for (int i = lo; i < hi; ++i) {
    rowstart[i] = run;
    cursor[i]   = run;
    run += deg[i];
  }
  if (lo < n && hi == n) rowstart[n] = run;
}

__global__ void k_scatter(const int* __restrict__ ei, int* __restrict__ cursor,
                          int* __restrict__ csr_src, int E, int total) {
  int e = blockIdx.x * blockDim.x + threadIdx.x;
  if (e >= total) return;
  int s, d; edge_sd(ei, e, E, s, d);
  int pos = atomicAdd(&cursor[d], 1);
  csr_src[pos] = s;
}

// ------------- segment softmax + weighted aggregation (gather) --------------
// one wave per (dst, head); no float atomics, outagg written exactly once.
__global__ void __launch_bounds__(256)
k_attn_aggregate(const int* __restrict__ rowstart, const int* __restrict__ csr_src,
                 const float* __restrict__ a_src, const float* __restrict__ a_dst,
                 const float* __restrict__ h, float* __restrict__ outagg,
                 int n_nodes) {
  const int wid  = (blockIdx.x * blockDim.x + threadIdx.x) >> 5;
  const int lane = threadIdx.x & 31;
  if (wid >= n_nodes * HEADS) return;
  const int dst = wid / HEADS;
  const int hd  = wid - dst * HEADS;

  const int start = rowstart[dst];
  const int end   = rowstart[dst + 1];
  const float ad  = a_dst[dst * HEADS + hd];

  // pass 1: segment max (lane-parallel over edges)
  float mx = -3.4e38f;
  for (int i = start + lane; i < end; i += 32) {
    int s = csr_src[i];
    mx = fmaxf(mx, lrelu(a_src[s * HEADS + hd] + ad));
  }
#pragma unroll
  for (int mask = 1; mask < 32; mask <<= 1) mx = fmaxf(mx, __shfl_xor(mx, mask));

  // pass 2: sum of exp
  float den = 0.f;
  for (int i = start + lane; i < end; i += 32) {
    int s = csr_src[i];
    den += __expf(lrelu(a_src[s * HEADS + hd] + ad) - mx);
  }
#pragma unroll
  for (int mask = 1; mask < 32; mask <<= 1) den += __shfl_xor(den, mask);

  // pass 3: accumulate sum_e exp(e) * h[s]   (lanes = channels, edges uniform)
  float acc0 = 0.f, acc1 = 0.f, acc2 = 0.f, acc3 = 0.f;
  for (int i = start; i < end; ++i) {
    int s = csr_src[i];
    float ex = __expf(lrelu(a_src[s * HEADS + hd] + ad) - mx);
    const float* hp = h + (size_t)s * HO + hd * OUTC;
    acc0 += ex * hp[lane];
    acc1 += ex * hp[lane + 32];
    acc2 += ex * hp[lane + 64];
    acc3 += ex * hp[lane + 96];
  }
  float inv = 1.0f / (den + 1e-16f);
  float* od = outagg + (size_t)dst * HO + hd * OUTC;
  od[lane]      = acc0 * inv;
  od[lane + 32] = acc1 * inv;
  od[lane + 64] = acc2 * inv;
  od[lane + 96] = acc3 * inv;
}

// ----------------------- relu+bias, global mean pool ------------------------
__global__ void k_pool(const float* __restrict__ outagg,
                       const float* __restrict__ bias,
                       const int* __restrict__ batch,
                       float* __restrict__ pooled, float* __restrict__ counts,
                       int n_nodes) {
  int tid = blockIdx.x * blockDim.x + threadIdx.x;
  if (tid >= n_nodes * HO) return;
  int n = tid / HO;
  int c = tid - n * HO;
  float v = outagg[tid] + bias[c];
  v = v > 0.f ? v : 0.f;
  int g = batch[n];
  atomicAdd(&pooled[g * HO + c], v);
  if (c == 0) atomicAdd(&counts[g], 1.0f);
}

// ------------------------------- final FC -----------------------------------
__global__ void k_head(const float* __restrict__ pooled,
                       const float* __restrict__ counts,
                       const float* __restrict__ fc_w,
                       const float* __restrict__ fc_b,
                       float* __restrict__ out) {
  int tid = threadIdx.x;                    // 128 = 64 graphs * 2 classes
  if (tid >= NUM_GRAPHS * NUM_CLASSES) return;
  int g   = tid / NUM_CLASSES;
  int cls = tid - g * NUM_CLASSES;
  float inv = 1.0f / fmaxf(counts[g], 1.0f);
  float s = 0.f;
  for (int k = 0; k < HO; ++k)
    s += pooled[g * HO + k] * inv * fc_w[k * NUM_CLASSES + cls];
  out[tid] = s + fc_b[cls];
}

// ---------------------------------------------------------------------------

static inline size_t align256(size_t v) { return (v + 255) & ~(size_t)255; }

extern "C" void kernel_launch(void* const* d_in, const int* in_sizes, int n_in,
                              void* d_out, int out_size, void* d_ws, size_t ws_size,
                              hipStream_t stream) {
  const float* x        = (const float*)d_in[0];
  const int*   ei       = (const int*)  d_in[1];
  const int*   batch    = (const int*)  d_in[2];
  const float* W        = (const float*)d_in[3];
  const float* att_src  = (const float*)d_in[4];
  const float* att_dst  = (const float*)d_in[5];
  const float* bias     = (const float*)d_in[6];
  const float* fc_w     = (const float*)d_in[7];
  const float* fc_b     = (const float*)d_in[8];
  float* out            = (float*)d_out;

  const int N  = in_sizes[0] / F_IN;      // 50000
  const int E  = in_sizes[1] / 2;         // 800000
  const int EN = E + N;                   // edges incl. self loops

  // workspace carve-up
  char* ws = (char*)d_ws;
  size_t off = 0;
  float*    h        = (float*)   (ws + off); off += align256((size_t)N * HO * 4);
  uint16_t* Wsw      = (uint16_t*)(ws + off); off += align256((size_t)8 * 24 * 32 * 16 * 2);
  float*    a_src    = (float*)   (ws + off); off += align256((size_t)N * HEADS * 4);
  float*    a_dst    = (float*)   (ws + off); off += align256((size_t)N * HEADS * 4);
  int*      deg      = (int*)     (ws + off); off += align256((size_t)N * 4);
  int*      rowstart = (int*)     (ws + off); off += align256((size_t)(N + 1) * 4);
  int*      cursor   = (int*)     (ws + off); off += align256((size_t)N * 4);
  int*      csr_src  = (int*)     (ws + off); off += align256((size_t)EN * 4);
  float*    outagg   = (float*)   (ws + off); off += align256((size_t)N * HO * 4);
  float*    pooled   = (float*)   (ws + off); off += align256((size_t)NUM_GRAPHS * HO * 4);
  float*    counts   = (float*)   (ws + off); off += align256((size_t)NUM_GRAPHS * 4);
  (void)ws_size; (void)n_in; (void)out_size;

  const int B = 256;
  auto nb = [](long long n, int b) { return (unsigned)((n + b - 1) / b); };

  // init accumulators (only small ones remain)
  k_fill_u32<<<nb(N, B), B, 0, stream>>>((uint32_t*)deg, 0u, N);
  k_fill_u32<<<nb(NUM_GRAPHS * HO, B), B, 0, stream>>>((uint32_t*)pooled, 0u, NUM_GRAPHS * HO);
  k_fill_u32<<<1, 64, 0, stream>>>((uint32_t*)counts, 0u, NUM_GRAPHS);

  // W -> bf16 WMMA-B-fragment order
  k_swizzle_W<<<nb(8 * 24 * 32 * 16, B), B, 0, stream>>>(W, Wsw);

  // h = x @ W (bf16 WMMA) with fused attention logits
  k_gemm_wmma<<<nb(N, 32), 192, 0, stream>>>(x, Wsw, att_src, att_dst,
                                             h, a_src, a_dst, N);

  // dst-CSR: histogram -> scan -> scatter
  k_deg<<<nb(EN, B), B, 0, stream>>>(ei, deg, E, EN);
  k_scan<<<1, 1024, 0, stream>>>(deg, rowstart, cursor, N);
  k_scatter<<<nb(EN, B), B, 0, stream>>>(ei, cursor, csr_src, E, EN);

  // segment softmax + aggregation: one wave per (dst, head)
  k_attn_aggregate<<<nb((long long)N * HEADS * 32, B), B, 0, stream>>>(
      rowstart, csr_src, a_src, a_dst, h, outagg, N);

  // relu + bias + global mean pool
  k_pool<<<nb((long long)N * HO, B), B, 0, stream>>>(outagg, bias, batch, pooled, counts, N);

  // final linear head
  k_head<<<1, 128, 0, stream>>>(pooled, counts, fc_w, fc_b, out);
}
